// MoELayer_66022237274826
// MI455X (gfx1250) — matrix operand order repaired and checked by reading, not compiled, not run on hardware
//
#include <hip/hip_runtime.h>

// ---------------------------------------------------------------------------
// MoE (top-2) for gfx1250: bf16 WMMA 16x16x32, fp32 accumulate.
// B=4, S=2048 -> N=8192 tokens, D=1024, E=8, F=2048.
// Expert kernel: 32 gathered tokens per block, X staged in LDS, H in LDS,
// B-fragments register double-buffered, 2x2 (N x M) WMMA blocking per wave.
// ---------------------------------------------------------------------------

#define D_DIM 1024
#define F_DIM 2048
#define E_NUM 8
#define TM    32           // tokens per workgroup in expert kernel
#define WAVES 8            // 256 threads

// LDS row strides (elements), padded +8 so successive rows rotate 4 banks:
// conflict-free 16-row ds_read_b128 fragment gathers.
#define XS_STRIDE (D_DIM + 8)    // 1032
#define HS_STRIDE (F_DIM + 8)    // 2056

typedef __attribute__((ext_vector_type(16))) __bf16 v16bf;
typedef __attribute__((ext_vector_type(8)))  __bf16 v8bf;
typedef __attribute__((ext_vector_type(8)))  float  v8f;

union BFrag {
    v16bf v;
    v8bf  h[2];
};

__device__ __forceinline__ v8f wmma_bf16(v16bf a, v16bf b, v8f c) {
    return __builtin_amdgcn_wmma_f32_16x16x32_bf16(
        /*neg_a=*/false, a, /*neg_b=*/false, b,
        /*c_mod=*/(short)0, c, /*reuse_a=*/false, /*reuse_b=*/false);
}

// Load a 16-value bf16 fragment as two b128 chunks; hi chunk at +hi_off elems.
// A- and B-operand: hi_off = 16
// (per ISA layout, a half-wave owns K chunks {0..7,16..23} / {8..15,24..31}).
__device__ __forceinline__ v16bf ldfrag(const __bf16* p, int hi_off) {
    BFrag f;
    f.h[0] = *(const v8bf*)(p);
    f.h[1] = *(const v8bf*)(p + hi_off);
    return f.v;
}

// -------------------------------- converts ---------------------------------

__global__ void cvt_bf16_kernel(const float* __restrict__ src,
                                __bf16* __restrict__ dst, int n) {
    int i = blockIdx.x * blockDim.x + threadIdx.x;
    int stride = gridDim.x * blockDim.x;
    for (; i < n; i += stride) dst[i] = (__bf16)src[i];
}

// src: [batch][R][C] f32  ->  dst: [batch][C][R] bf16 (32x32 LDS tile transpose)
__global__ void transpose_cvt_kernel(const float* __restrict__ src,
                                     __bf16* __restrict__ dst, int R, int C) {
    __shared__ float tile[32][33];
    const float* s = src + (size_t)blockIdx.z * R * C;
    __bf16*      d = dst + (size_t)blockIdx.z * R * C;
    int c0 = blockIdx.x * 32, r0 = blockIdx.y * 32;
    int tx = threadIdx.x, ty = threadIdx.y;   // 32 x 8
    #pragma unroll
    for (int i = 0; i < 32; i += 8)
        tile[ty + i][tx] = s[(size_t)(r0 + ty + i) * C + c0 + tx];
    __syncthreads();
    #pragma unroll
    for (int i = 0; i < 32; i += 8)
        d[(size_t)(c0 + ty + i) * R + r0 + tx] = (__bf16)tile[tx][ty + i];
}

// --------------------------------- router ----------------------------------

__global__ void router_kernel(const float* __restrict__ x,
                              const int* __restrict__ mask,
                              const float* __restrict__ rw,   // [D,E]
                              int N,
                              int* __restrict__ te,           // [2N] expert ids (-1 masked)
                              float* __restrict__ tw,         // [2N] weights
                              int* __restrict__ counts) {     // [E]
    int token = blockIdx.x * (blockDim.x >> 5) + (threadIdx.x >> 5);
    int lane = threadIdx.x & 31;
    if (token >= N) return;

    float acc[E_NUM];
    #pragma unroll
    for (int e = 0; e < E_NUM; ++e) acc[e] = 0.0f;

    const float* xr = x + (size_t)token * D_DIM;
    for (int d = lane; d < D_DIM; d += 32) {
        float xv = xr[d];
        const float* r = rw + (size_t)d * E_NUM;
        #pragma unroll
        for (int e = 0; e < E_NUM; ++e) acc[e] += xv * r[e];
    }
    #pragma unroll
    for (int e = 0; e < E_NUM; ++e) {
        #pragma unroll
        for (int off = 16; off >= 1; off >>= 1)
            acc[e] += __shfl_xor(acc[e], off, 32);
    }
    if (lane == 0) {
        float mx = acc[0];
        #pragma unroll
        for (int e = 1; e < E_NUM; ++e) mx = fmaxf(mx, acc[e]);
        float p[E_NUM];
        #pragma unroll
        for (int e = 0; e < E_NUM; ++e) p[e] = __expf(acc[e] - mx);
        int i0 = 0;
        #pragma unroll
        for (int e = 1; e < E_NUM; ++e) if (p[e] > p[i0]) i0 = e;
        int i1 = (i0 == 0) ? 1 : 0;
        #pragma unroll
        for (int e = 0; e < E_NUM; ++e)
            if (e != i0 && p[e] > p[i1]) i1 = e;
        float inv = 1.0f / (p[i0] + p[i1]);
        bool ok = (mask[token] != 0);
        te[2 * token]     = ok ? i0 : -1;
        te[2 * token + 1] = ok ? i1 : -1;
        tw[2 * token]     = p[i0] * inv;
        tw[2 * token + 1] = p[i1] * inv;
        if (ok) {
            atomicAdd(&counts[i0], 1);
            atomicAdd(&counts[i1], 1);
        }
    }
}

__global__ void scan_kernel(const int* __restrict__ counts,
                            int* __restrict__ offsets,
                            int* __restrict__ cursors) {
    if (threadIdx.x == 0 && blockIdx.x == 0) {
        int s = 0;
        #pragma unroll
        for (int e = 0; e < E_NUM; ++e) {
            offsets[e] = s;
            s += counts[e];
            cursors[e] = 0;
        }
        offsets[E_NUM] = s;
    }
}

__global__ void scatter_kernel(const int* __restrict__ te,
                               const float* __restrict__ tw,
                               const int* __restrict__ offsets,
                               int* __restrict__ cursors,
                               int* __restrict__ btok,
                               float* __restrict__ bwgt, int A) {
    int i = blockIdx.x * blockDim.x + threadIdx.x;
    if (i >= A) return;
    int e = te[i];
    if (e < 0) return;
    int pos = atomicAdd(&cursors[e], 1);
    int idx = offsets[e] + pos;
    btok[idx] = i >> 1;
    bwgt[idx] = tw[i];
}

// --------------------------- fused expert kernel ----------------------------
// grid = (maxTiles, E). 32 gathered token rows:
//   stage X[32,1024]bf16 in LDS -> H = relu(X@W1+b1) bf16 in LDS
//   -> Y = H@W2+b2 -> out += wgt*Y (f32 atomics, 2 commutative adds/element).

__global__ void __launch_bounds__(256, 2)
moe_expert_kernel(const __bf16* __restrict__ xb,   // [N,D] bf16
                  const __bf16* __restrict__ w1t,  // [E,F,D] bf16 (K=d contiguous)
                  const float*  __restrict__ b1,   // [E,F]
                  const __bf16* __restrict__ w2t,  // [E,D,F] bf16 (K=f contiguous)
                  const float*  __restrict__ b2,   // [E,D]
                  const int*    __restrict__ offsets,
                  const int*    __restrict__ btok,
                  const float*  __restrict__ bwgt,
                  float* __restrict__ out) {       // [N,D]
    extern __shared__ unsigned char smem_raw[];
    int*    toks = (int*)smem_raw;                 // 32 ints
    float*  wg   = (float*)(smem_raw + 128);       // 32 floats
    __bf16* Xs   = (__bf16*)(smem_raw + 512);                        // [32][XS_STRIDE]
    __bf16* Hs   = (__bf16*)(smem_raw + 512 + TM * XS_STRIDE * 2);   // [32][HS_STRIDE]

    const int e   = blockIdx.y;
    const int seg = offsets[e];
    const int n_e = offsets[e + 1] - seg;
    const int m0  = blockIdx.x * TM;
    if (m0 >= n_e) return;
    int rows = n_e - m0;
    if (rows > TM) rows = TM;

    const int tid = threadIdx.x;
    if (tid < TM) {
        toks[tid] = (tid < rows) ? btok[seg + m0 + tid] : 0;
        wg[tid]   = (tid < rows) ? bwgt[seg + m0 + tid] : 0.0f;
    }
    __syncthreads();

    // ---- stage gathered X tile into LDS (coalesced 16B chunks) -------------
    {
        int r   = tid >> 3;                    // 32 rows, 8 threads per row
        int sub = tid & 7;
        const __bf16* xg = xb + (size_t)toks[r] * D_DIM;
        __bf16*       xd = Xs + r * XS_STRIDE;
        #pragma unroll
        for (int it = 0; it < D_DIM / (8 * 8); ++it) {
            int c = (sub + it * 8) * 8;
            *(v8bf*)(xd + c) = *(const v8bf*)(xg + c);
        }
    }
    __syncthreads();

    const int wave = tid >> 5;
    const int lane = tid & 31;
    const int half = lane >> 4;
    const int l16  = lane & 31 & 15;

    const __bf16* W1e = w1t + (size_t)e * F_DIM * D_DIM;
    const __bf16* W2e = w2t + (size_t)e * D_DIM * F_DIM;

    const __bf16* a0base = Xs + (0 * 16 + l16) * XS_STRIDE + half * 8;
    const __bf16* a1base = Xs + (1 * 16 + l16) * XS_STRIDE + half * 8;

    // ---------------- GEMM1: H = relu(X @ W1 + b1) -> LDS -------------------
    // 64 n-blocks of 32 cols; each wave takes every 8th block (2x2 WMMA tile).
    #pragma unroll 1
    for (int b = wave; b < F_DIM / 32; b += WAVES) {
        const int cb = b * 32;
        const __bf16* b0p = W1e + (size_t)(cb + l16) * D_DIM + half * 8;
        const __bf16* b1p = b0p + (size_t)16 * D_DIM;

        v16bf cB0 = ldfrag(b0p, 16), cB1 = ldfrag(b1p, 16);
        v8f acc00 = {}, acc01 = {}, acc10 = {}, acc11 = {};

        #pragma unroll 2
        for (int s = 0; s < D_DIM / 32 - 1; ++s) {
            const int k = s * 32;
            v16bf nB0 = ldfrag(b0p + k + 32, 16);
            v16bf nB1 = ldfrag(b1p + k + 32, 16);
            v16bf A0  = ldfrag(a0base + k, 16);
            v16bf A1  = ldfrag(a1base + k, 16);
            acc00 = wmma_bf16(A0, cB0, acc00);
            acc10 = wmma_bf16(A1, cB0, acc10);
            acc01 = wmma_bf16(A0, cB1, acc01);
            acc11 = wmma_bf16(A1, cB1, acc11);
            cB0 = nB0; cB1 = nB1;
        }
        {
            const int k = D_DIM - 32;
            v16bf A0 = ldfrag(a0base + k, 16);
            v16bf A1 = ldfrag(a1base + k, 16);
            acc00 = wmma_bf16(A0, cB0, acc00);
            acc10 = wmma_bf16(A1, cB0, acc10);
            acc01 = wmma_bf16(A0, cB1, acc01);
            acc11 = wmma_bf16(A1, cB1, acc11);
        }

        float bias0 = b1[e * F_DIM + cb + l16];
        float bias1 = b1[e * F_DIM + cb + 16 + l16];
        #pragma unroll
        for (int j = 0; j < 8; ++j) {
            int r0 = j + 8 * half;         // tile rows 0..15
            int r1 = 16 + r0;              // tile rows 16..31
            float v;
            v = acc00[j] + bias0; Hs[r0 * HS_STRIDE + cb + l16]      = (__bf16)(v > 0.f ? v : 0.f);
            v = acc10[j] + bias0; Hs[r1 * HS_STRIDE + cb + l16]      = (__bf16)(v > 0.f ? v : 0.f);
            v = acc01[j] + bias1; Hs[r0 * HS_STRIDE + cb + 16 + l16] = (__bf16)(v > 0.f ? v : 0.f);
            v = acc11[j] + bias1; Hs[r1 * HS_STRIDE + cb + 16 + l16] = (__bf16)(v > 0.f ? v : 0.f);
        }
    }
    __syncthreads();

    const __bf16* h0base = Hs + (0 * 16 + l16) * HS_STRIDE + half * 8;
    const __bf16* h1base = Hs + (1 * 16 + l16) * HS_STRIDE + half * 8;

    // ------------- GEMM2: Y = H @ W2 + b2 ; out += wgt * Y ------------------
    #pragma unroll 1
    for (int b = wave; b < D_DIM / 32; b += WAVES) {
        const int cb = b * 32;
        const __bf16* b0p = W2e + (size_t)(cb + l16) * F_DIM + half * 8;
        const __bf16* b1p = b0p + (size_t)16 * F_DIM;

        v16bf cB0 = ldfrag(b0p, 16), cB1 = ldfrag(b1p, 16);
        v8f acc00 = {}, acc01 = {}, acc10 = {}, acc11 = {};

        #pragma unroll 2
        for (int s = 0; s < F_DIM / 32 - 1; ++s) {
            const int k = s * 32;
            v16bf nB0 = ldfrag(b0p + k + 32, 16);
            v16bf nB1 = ldfrag(b1p + k + 32, 16);
            v16bf A0  = ldfrag(h0base + k, 16);
            v16bf A1  = ldfrag(h1base + k, 16);
            acc00 = wmma_bf16(A0, cB0, acc00);
            acc10 = wmma_bf16(A1, cB0, acc10);
            acc01 = wmma_bf16(A0, cB1, acc01);
            acc11 = wmma_bf16(A1, cB1, acc11);
            cB0 = nB0; cB1 = nB1;
        }
        {
            const int k = F_DIM - 32;
            v16bf A0 = ldfrag(h0base + k, 16);
            v16bf A1 = ldfrag(h1base + k, 16);
            acc00 = wmma_bf16(A0, cB0, acc00);
            acc10 = wmma_bf16(A1, cB0, acc10);
            acc01 = wmma_bf16(A0, cB1, acc01);
            acc11 = wmma_bf16(A1, cB1, acc11);
        }

        float bias0 = b2[e * D_DIM + cb + l16];
        float bias1 = b2[e * D_DIM + cb + 16 + l16];
        int col0 = cb + l16, col1 = cb + 16 + l16;
        #pragma unroll
        for (int j = 0; j < 8; ++j) {
            int r0 = j + 8 * half;
            int r1 = 16 + r0;
            if (r0 < rows) {
                atomicAdd(&out[(size_t)toks[r0] * D_DIM + col0], (acc00[j] + bias0) * wg[r0]);
                atomicAdd(&out[(size_t)toks[r0] * D_DIM + col1], (acc01[j] + bias1) * wg[r0]);
            }
            if (r1 < rows) {
                atomicAdd(&out[(size_t)toks[r1] * D_DIM + col0], (acc10[j] + bias0) * wg[r1]);
                atomicAdd(&out[(size_t)toks[r1] * D_DIM + col1], (acc11[j] + bias1) * wg[r1]);
            }
        }
    }
}

// ------------------------------- launcher -----------------------------------

extern "C" void kernel_launch(void* const* d_in, const int* in_sizes, int n_in,
                              void* d_out, int out_size, void* d_ws, size_t ws_size,
                              hipStream_t stream) {
    const float* x    = (const float*)d_in[0];   // [B,S,D]
    const int*   mask = (const int*)d_in[1];     // [B,S]
    const float* rw   = (const float*)d_in[2];   // [D,E]
    const float* w1   = (const float*)d_in[3];   // [E,D,F]
    const float* b1   = (const float*)d_in[4];   // [E,F]
    const float* w2   = (const float*)d_in[5];   // [E,F,D]
    const float* b2   = (const float*)d_in[6];   // [E,D]
    float* out = (float*)d_out;

    const int N = in_sizes[0] / D_DIM;           // B*S = 8192 tokens
    const int A = 2 * N;                         // assignments (top-2)

    char* ws = (char*)d_ws;
    size_t off = 0;
    auto carve = [&](size_t bytes) -> void* {
        void* p = ws + off;
        off = (off + bytes + 255) & ~(size_t)255;
        return p;
    };
    __bf16* xb   = (__bf16*)carve((size_t)N * D_DIM * 2);
    __bf16* w1t  = (__bf16*)carve((size_t)E_NUM * D_DIM * F_DIM * 2);
    __bf16* w2t  = (__bf16*)carve((size_t)E_NUM * D_DIM * F_DIM * 2);
    int*    te   = (int*)carve((size_t)A * 4);
    float*  tw   = (float*)carve((size_t)A * 4);
    int*    cnts = (int*)carve(E_NUM * 4);
    int*    offs = (int*)carve((E_NUM + 1) * 4);
    int*    curs = (int*)carve(E_NUM * 4);
    int*    btok = (int*)carve((size_t)A * 4);
    float*  bwgt = (float*)carve((size_t)A * 4);
    (void)ws_size; (void)n_in;

    hipMemsetAsync(cnts, 0, E_NUM * 4, stream);
    hipMemsetAsync(out, 0, (size_t)out_size * sizeof(float), stream);

    // 1) fp32 -> bf16 hidden states
    cvt_bf16_kernel<<<2048, 256, 0, stream>>>(x, xb, N * D_DIM);

    // 2) transpose+convert weights: w1 [E,D,F]->[E,F,D]; w2 [E,F,D]->[E,D,F]
    {
        dim3 blk(32, 8);
        dim3 g1(F_DIM / 32, D_DIM / 32, E_NUM);
        transpose_cvt_kernel<<<g1, blk, 0, stream>>>(w1, w1t, D_DIM, F_DIM);
        dim3 g2(D_DIM / 32, F_DIM / 32, E_NUM);
        transpose_cvt_kernel<<<g2, blk, 0, stream>>>(w2, w2t, F_DIM, D_DIM);
    }
    // 3) router (wave per token) + histogram
    {
        int tok_per_blk = 256 / 32;
        router_kernel<<<(N + tok_per_blk - 1) / tok_per_blk, 256, 0, stream>>>(
            x, mask, rw, N, te, tw, cnts);
    }
    // 4) scan + scatter into per-expert buckets
    scan_kernel<<<1, 32, 0, stream>>>(cnts, offs, curs);
    scatter_kernel<<<(A + 255) / 256, 256, 0, stream>>>(te, tw, offs, curs,
                                                        btok, bwgt, A);
    // 5) fused top-2 expert MLP with WMMA
    {
        int maxTiles = (A + TM - 1) / TM;        // 512
        dim3 grid(maxTiles, E_NUM);
        size_t smem = 512 + (size_t)TM * XS_STRIDE * 2 + (size_t)TM * HS_STRIDE * 2;
        moe_expert_kernel<<<grid, WAVES * 32, smem, stream>>>(
            xb, w1t, b1, w2t, b2, offs, btok, bwgt, out);
    }
}
